// Attention_61959198212160
// MI455X (gfx1250) — matrix operand order repaired and checked by reading, not compiled, not run on hardware
//
#include <hip/hip_runtime.h>
#include <math.h>
#include <stdint.h>

// ---------------------------------------------------------------------------
// Problem constants (match reference)
// ---------------------------------------------------------------------------
#define DE 1024          // d_embed
#define NH 16            // heads
#define DKH 64           // d_k per head
#define BB 4             // batch
#define SSEQ 2048        // sequence
#define MS (BB * SSEQ)   // 8192 total rows

typedef __attribute__((ext_vector_type(16))) __bf16 v16bf;
typedef __attribute__((ext_vector_type(8)))  __bf16 v8bf;
typedef __attribute__((ext_vector_type(8)))  float  v8f;
typedef __attribute__((ext_vector_type(4)))  int    v4i;

// ---------------------------------------------------------------------------
// gfx1250 async global->LDS copy (ASYNCcnt path), with portable fallback
// ---------------------------------------------------------------------------
#if defined(__has_builtin)
#  if __has_builtin(__builtin_amdgcn_global_load_async_to_lds_b128)
#    define HAVE_ASYNC_LDS 1
#  endif
#endif
#ifndef HAVE_ASYNC_LDS
#  define HAVE_ASYNC_LDS 0
#endif

typedef __attribute__((address_space(1))) v4i GAS_v4i;
typedef __attribute__((address_space(3))) v4i LAS_v4i;

__device__ __forceinline__ void cp_b128_to_lds(const void* g, void* l) {
#if HAVE_ASYNC_LDS
  __builtin_amdgcn_global_load_async_to_lds_b128((GAS_v4i*)g, (LAS_v4i*)l, 0, 0);
#else
  *(v4i*)l = *(const v4i*)g;   // synchronous fallback: global load + ds store
#endif
}

__device__ __forceinline__ void cp_lds_wait() {
#if HAVE_ASYNC_LDS
#  if __has_builtin(__builtin_amdgcn_s_wait_asynccnt)
  __builtin_amdgcn_s_wait_asynccnt(0);
#  else
  asm volatile("s_wait_asynccnt 0" ::: "memory");
#  endif
#endif
}

// ---------------------------------------------------------------------------
// WMMA helpers (CDNA5 wave32, V_WMMA_F32_16X16X32_BF16)
// ---------------------------------------------------------------------------
__device__ __forceinline__ v8f wmma_bf16(v16bf a, v16bf b, v8f c) {
  return __builtin_amdgcn_wmma_f32_16x16x32_bf16(
      /*neg_a=*/false, a, /*neg_b=*/false, b,
      /*c_mod=*/(short)0, c, /*reuse_a=*/false, /*reuse_b=*/false);
}

// A-matrix fragment (16x32 bf16, MxK). Contiguous k order (same as frag_b): lane L row = L%16,
// elements i=0..15 -> K = off+i,
// off = (L>=16)*16.  base points at element [row0, k0] of a row-major matrix.
__device__ __forceinline__ v16bf frag_a(const __bf16* base, int ld) {
  const int lane = threadIdx.x & 31;
  const __bf16* p = base + (lane & 15) * ld + ((lane >> 4) << 4);
  v8bf c0 = *reinterpret_cast<const v8bf*>(p);
  v8bf c1 = *reinterpret_cast<const v8bf*>(p + 8);
  v16bf r;
#pragma unroll
  for (int i = 0; i < 8; ++i) { r[i] = c0[i]; r[i + 8] = c1[i]; }
  return r;
}

// B-matrix fragment (32x16 bf16, KxN). Contiguous k order (same as frag_a): lane L col = L%16,
// elements i=0..15 -> K = koff+i, koff = (L>=16)*16.  "base" points at the
// matrix whose ROW n is column n of B (i.e. B^T stored row-major, ld given).
__device__ __forceinline__ v16bf frag_b(const __bf16* base, int ld) {
  const int lane = threadIdx.x & 31;
  const __bf16* p = base + (lane & 15) * ld + ((lane >> 4) << 4);
  return *reinterpret_cast<const v16bf*>(p);
}

// ---------------------------------------------------------------------------
// Kernel 0: fp32 -> bf16 conversion (grid-stride)
// ---------------------------------------------------------------------------
__global__ void cvt_f32_bf16(const float* __restrict__ in, __bf16* __restrict__ out,
                             size_t n) {
  size_t i = (size_t)blockIdx.x * blockDim.x + threadIdx.x;
  size_t stride = (size_t)gridDim.x * blockDim.x;
  for (; i < n; i += stride) out[i] = (__bf16)in[i];
}

// ---------------------------------------------------------------------------
// Kernel 1/3: WMMA GEMM  C[M,N] = A[M,K] @ W[N,K]^T + bias[N]
//   Workgroup (8 waves, 256 thr) computes a 128x64 tile; the shared 64xK W
//   strip is staged through double-buffered LDS with async global->LDS
//   copies so all 8 waves read W once per block.
//   mode 0: store bf16 row-major            (Q, K projections)
//   mode 1: store bf16 transposed per-head  (V -> Vt[(b*DE + n)*S + s])
//   mode 2: store fp32 row-major            (output projection)
// ---------------------------------------------------------------------------
__global__ void gemm_wmma(const __bf16* __restrict__ A, const __bf16* __restrict__ W,
                          const float* __restrict__ bias,
                          __bf16* __restrict__ outb, float* __restrict__ outf,
                          int M, int N, int Kd, int mode, int Sdim) {
  __shared__ __align__(16) __bf16 wtile[2][64][32];   // 8 KiB, double-buffered

  const int tiles_n = N >> 6;
  const int bm = blockIdx.x / tiles_n;                // 128-row macro tile
  const int bn = blockIdx.x % tiles_n;                // 64-col strip
  const int wave = threadIdx.x >> 5;
  const int lane = threadIdx.x & 31;
  const int row0 = bm * 128 + wave * 16;

  // cooperative W staging: thread t copies 16B: row ln, element chunk ch
  const int ln = threadIdx.x >> 2;                    // 0..63
  const int ch = (threadIdx.x & 3) * 8;               // 0,8,16,24 (bf16 elems)
  const __bf16* wsrc = W + (size_t)(bn * 64 + ln) * Kd + ch;

  cp_b128_to_lds(wsrc, &wtile[0][ln][ch]);            // prologue: k-slice 0
  cp_lds_wait();
  __syncthreads();

  v8f acc[4] = {};
  const __bf16* Abase = A + (size_t)row0 * Kd;
  int cur = 0;
  for (int k0 = 0; k0 < Kd; k0 += 32) {
    if (k0 + 32 < Kd)                                 // prefetch next k-slice
      cp_b128_to_lds(wsrc + k0 + 32, &wtile[cur ^ 1][ln][ch]);

    v16bf af = frag_a(Abase + k0, Kd);
#pragma unroll
    for (int nt = 0; nt < 4; ++nt)
      acc[nt] = wmma_bf16(af, frag_b(&wtile[cur][nt * 16][0], 32), acc[nt]);

    cp_lds_wait();                                    // prefetch landed
    __syncthreads();                                  // visible to all waves
    cur ^= 1;
  }

  const int col = lane & 15;
  const int rb  = (lane >> 4) * 8;
#pragma unroll
  for (int nt = 0; nt < 4; ++nt) {
#pragma unroll
    for (int r = 0; r < 8; ++r) {
      const int m = row0 + rb + r;
      const int n = bn * 64 + nt * 16 + col;
      float v = acc[nt][r] + bias[n];
      if (mode == 0) {
        outb[(size_t)m * N + n] = (__bf16)v;
      } else if (mode == 1) {
        const int bi = m / Sdim, s = m % Sdim;
        outb[((size_t)bi * N + n) * Sdim + s] = (__bf16)v;   // Vt layout
      } else {
        outf[(size_t)m * N + n] = v;
      }
    }
  }
}

// ---------------------------------------------------------------------------
// Kernel 2: causal flash attention, one wave per (b, h, 16-query tile).
//   Q,K bf16 row-major [MS, DE]; Vt bf16 [(b*DE + n), SSEQ]; ctx bf16 [MS, DE]
// ---------------------------------------------------------------------------
__global__ void attn_wmma(const __bf16* __restrict__ Q, const __bf16* __restrict__ K,
                          const __bf16* __restrict__ Vt, __bf16* __restrict__ ctx) {
  __shared__ __align__(16) float  sc[16][64];
  __shared__ __align__(16) __bf16 pbuf[16][64];
  __shared__ float rowmax[16], rowsum[16], ralpha[16];

  const int lane = threadIdx.x;                // 32 threads
  const int qt_per = SSEQ / 16;                // 128
  const int b  = blockIdx.x / (NH * qt_per);
  const int rm = blockIdx.x % (NH * qt_per);
  const int h  = rm / qt_per;
  const int qt = rm % qt_per;
  const int qbase = qt * 16;

  const __bf16* Qrow = Q + ((size_t)(b * SSEQ + qbase)) * DE + h * DKH;
  const v16bf qf0 = frag_a(Qrow, DE);
  const v16bf qf1 = frag_a(Qrow + 32, DE);

  v8f acc[4] = {};
  if (lane < 16) { rowmax[lane] = -__builtin_inff(); rowsum[lane] = 0.f; }
  __syncthreads();

  const int col = lane & 15;
  const int rb  = (lane >> 4) * 8;

  for (int kbase = 0; kbase <= qbase + 15; kbase += 64) {
    // ---- scores: S = (Q K^T) / sqrt(dk), causal mask, to LDS ----
#pragma unroll
    for (int nt = 0; nt < 4; ++nt) {
      const __bf16* Krow = K + ((size_t)(b * SSEQ + kbase + nt * 16)) * DE + h * DKH;
      v8f s = {};
      s = wmma_bf16(qf0, frag_b(Krow, DE), s);
      s = wmma_bf16(qf1, frag_b(Krow + 32, DE), s);
      const int n = nt * 16 + col;
#pragma unroll
      for (int r = 0; r < 8; ++r) {
        const int m = rb + r;
        float v = s[r] * 0.125f;                       // 1/sqrt(64)
        if (kbase + n > qbase + m) v = -__builtin_inff();
        sc[m][n] = v;
      }
    }
    __syncthreads();

    // ---- online softmax (one lane per query row) ----
    if (lane < 16) {
      const int m = lane;
      float bm = -__builtin_inff();
      for (int j = 0; j < 64; ++j) bm = fmaxf(bm, sc[m][j]);
      const float om = rowmax[m];
      const float nm = fmaxf(om, bm);
      const float al = expf(om - nm);                  // 0 on first block, 1 when bm masked
      float bs = 0.f;
      for (int j = 0; j < 64; ++j) {
        float p = expf(sc[m][j] - nm);
        pbuf[m][j] = (__bf16)p;
        bs += p;
      }
      rowsum[m] = rowsum[m] * al + bs;
      rowmax[m] = nm;
      ralpha[m] = al;
    }
    __syncthreads();

    // ---- rescale accumulators, then ctx += P @ V ----
    const v16bf p0 = frag_a(&pbuf[0][0], 64);
    const v16bf p1 = frag_a(&pbuf[0][0] + 32, 64);
#pragma unroll
    for (int nt = 0; nt < 4; ++nt) {
#pragma unroll
      for (int r = 0; r < 8; ++r) acc[nt][r] *= ralpha[rb + r];
      const __bf16* Vrow = Vt + ((size_t)(b * DE + h * DKH + nt * 16)) * SSEQ + kbase;
      acc[nt] = wmma_bf16(p0, frag_b(Vrow, SSEQ), acc[nt]);
      acc[nt] = wmma_bf16(p1, frag_b(Vrow + 32, SSEQ), acc[nt]);
    }
    __syncthreads();
  }

  // ---- epilogue: ctx = acc / rowsum, store bf16 row-major ----
#pragma unroll
  for (int nt = 0; nt < 4; ++nt) {
#pragma unroll
    for (int r = 0; r < 8; ++r) {
      const int m = rb + r;
      const float v = acc[nt][r] / rowsum[m];
      ctx[((size_t)(b * SSEQ + qbase + m)) * DE + h * DKH + nt * 16 + col] = (__bf16)v;
    }
  }
}

// ---------------------------------------------------------------------------
// Kernel 4: out = x + LayerNorm(proj) * gamma + beta   (one block per row)
// ---------------------------------------------------------------------------
__global__ void ln_residual(const float* __restrict__ x, const float* __restrict__ proj,
                            const float* __restrict__ gamma, const float* __restrict__ beta,
                            float* __restrict__ out) {
  const int row = blockIdx.x;
  const float* p  = proj + (size_t)row * DE;
  const float* xr = x    + (size_t)row * DE;

  float s = 0.f, s2 = 0.f;
  for (int i = threadIdx.x; i < DE; i += blockDim.x) {
    const float v = p[i];
    s += v; s2 += v * v;
  }
#pragma unroll
  for (int o = 16; o > 0; o >>= 1) {
    s  += __shfl_down(s,  o, 32);
    s2 += __shfl_down(s2, o, 32);
  }
  __shared__ float sa[8], sb[8];
  __shared__ float smu, sinv;
  const int w = threadIdx.x >> 5;
  if ((threadIdx.x & 31) == 0) { sa[w] = s; sb[w] = s2; }
  __syncthreads();
  if (threadIdx.x == 0) {
    float a = 0.f, b2 = 0.f;
    const int nw = (int)(blockDim.x >> 5);
    for (int i = 0; i < nw; ++i) { a += sa[i]; b2 += sb[i]; }
    const float mu = a / (float)DE;
    smu  = mu;
    sinv = rsqrtf(b2 / (float)DE - mu * mu + 1e-5f);
  }
  __syncthreads();
  const float mu = smu, inv = sinv;
  for (int i = threadIdx.x; i < DE; i += blockDim.x)
    out[(size_t)row * DE + i] = xr[i] + (p[i] - mu) * inv * gamma[i] + beta[i];
}

// ---------------------------------------------------------------------------
// Host launcher
// ---------------------------------------------------------------------------
extern "C" void kernel_launch(void* const* d_in, const int* in_sizes, int n_in,
                              void* d_out, int out_size, void* d_ws, size_t ws_size,
                              hipStream_t stream) {
  (void)in_sizes; (void)n_in; (void)out_size; (void)ws_size;
  const float* x     = (const float*)d_in[0];
  const float* Wq    = (const float*)d_in[1];
  const float* bq    = (const float*)d_in[2];
  const float* Wk    = (const float*)d_in[3];
  const float* bk    = (const float*)d_in[4];
  const float* Wv    = (const float*)d_in[5];
  const float* bv    = (const float*)d_in[6];
  const float* Wo    = (const float*)d_in[7];
  const float* bo    = (const float*)d_in[8];
  const float* gamma = (const float*)d_in[9];
  const float* beta  = (const float*)d_in[10];
  float* out = (float*)d_out;

  const size_t nx = (size_t)MS * DE;   // 8,388,608
  const size_t nw = (size_t)DE * DE;   // 1,048,576

  char* ws = (char*)d_ws;
  __bf16* xb   = (__bf16*)ws;                      ws += nx * sizeof(__bf16);
  __bf16* wqb  = (__bf16*)ws;                      ws += nw * sizeof(__bf16);
  __bf16* wkb  = (__bf16*)ws;                      ws += nw * sizeof(__bf16);
  __bf16* wvb  = (__bf16*)ws;                      ws += nw * sizeof(__bf16);
  __bf16* wob  = (__bf16*)ws;                      ws += nw * sizeof(__bf16);
  __bf16* qb   = (__bf16*)ws;                      ws += nx * sizeof(__bf16);
  __bf16* kbuf = (__bf16*)ws;                      ws += nx * sizeof(__bf16);
  __bf16* vtb  = (__bf16*)ws;                      ws += nx * sizeof(__bf16);
  __bf16* ctxb = (__bf16*)ws;                      ws += nx * sizeof(__bf16);
  float*  projf = (float*)ws;                      // nx floats

  // 0) convert inputs to bf16
  cvt_f32_bf16<<<2048, 256, 0, stream>>>(x,  xb,  nx);
  cvt_f32_bf16<<<512,  256, 0, stream>>>(Wq, wqb, nw);
  cvt_f32_bf16<<<512,  256, 0, stream>>>(Wk, wkb, nw);
  cvt_f32_bf16<<<512,  256, 0, stream>>>(Wv, wvb, nw);
  cvt_f32_bf16<<<512,  256, 0, stream>>>(Wo, wob, nw);

  // 1) Q, K, V projections: 128x64 macro tiles, 8 waves/block
  const int gemm_blocks = (MS / 128) * (DE / 64);      // 1024
  gemm_wmma<<<gemm_blocks, 256, 0, stream>>>(xb, wqb, bq, qb,   nullptr, MS, DE, DE, 0, SSEQ);
  gemm_wmma<<<gemm_blocks, 256, 0, stream>>>(xb, wkb, bk, kbuf, nullptr, MS, DE, DE, 0, SSEQ);
  gemm_wmma<<<gemm_blocks, 256, 0, stream>>>(xb, wvb, bv, vtb,  nullptr, MS, DE, DE, 1, SSEQ);

  // 2) causal attention: one wave per (b, h, 16-query tile)
  attn_wmma<<<BB * NH * (SSEQ / 16), 32, 0, stream>>>(qb, kbuf, vtb, ctxb);

  // 3) output projection (fp32 out + bias)
  gemm_wmma<<<gemm_blocks, 256, 0, stream>>>(ctxb, wob, bo, nullptr, projf, MS, DE, DE, 2, SSEQ);

  // 4) residual + layernorm
  ln_residual<<<MS, 256, 0, stream>>>(x, projf, gamma, beta, out);
}
